// MultiHeadedAttention_29661044146475
// MI455X (gfx1250) — matrix operand order repaired and checked by reading, not compiled, not run on hardware
//
#include <hip/hip_runtime.h>

// ---------------- types ----------------
typedef __bf16 bf16;
typedef __attribute__((ext_vector_type(8)))  __bf16 v8bf;
typedef __attribute__((ext_vector_type(16))) __bf16 v16bf;
typedef __attribute__((ext_vector_type(8)))  float  v8f;

#define NB_B   2
#define NB_S   2048
#define NB_D   1024
#define NB_H   16
#define NB_DK  64
#define CHUNK  512           // key-columns accumulated in LDS per pass

__device__ __forceinline__ v8f zero8() {
  v8f z;
#pragma unroll
  for (int i = 0; i < 8; ++i) z[i] = 0.0f;
  return z;
}

// load two contiguous 16B chunks into one 16-elem bf16 fragment
__device__ __forceinline__ v16bf ld_pair(const bf16* p0, const bf16* p1) {
  v8bf a = *(const v8bf*)p0;
  v8bf b = *(const v8bf*)p1;
  v16bf r;
#pragma unroll
  for (int i = 0; i < 8; ++i) { r[i] = a[i]; r[i + 8] = b[i]; }
  return r;
}

__device__ __forceinline__ v8f wmma_bf16(v16bf a, v16bf b, v8f c) {
  // D = A(16x32 bf16) * B(32x16 bf16) + C(16x16 f32)
  return __builtin_amdgcn_wmma_f32_16x16x32_bf16(false, a, false, b, (short)0, c,
                                                 false, false);
}

// ---------------- fp32 -> bf16 convert ----------------
__global__ __launch_bounds__(256) void cvt_f32_to_bf16(const float* __restrict__ in,
                                                       bf16* __restrict__ out, int n) {
  int i = (blockIdx.x * 256 + threadIdx.x) * 4;
  if (i < n) {
    float4 f = *(const float4*)(in + i);
    out[i + 0] = (bf16)f.x;
    out[i + 1] = (bf16)f.y;
    out[i + 2] = (bf16)f.z;
    out[i + 3] = (bf16)f.w;
  }
}

// ---------------- WMMA GEMM: C = alpha * (A[MxK] @ W[NxK]^T + bias) ----------------
// MODE 0: row-major OutT output [M,N]
// MODE 1: bf16 output transposed per batch: out[b][n][t] with m = b*Stok + t
template <int MODE, typename OutT>
__global__ __launch_bounds__(256) void gemm_wmma_kernel(
    const bf16* __restrict__ A, const bf16* __restrict__ W,
    const float* __restrict__ bias, OutT* __restrict__ Cout,
    int M, int N, int K, int Stok, float alpha) {
  const int lane = threadIdx.x & 31;
  const int wave = threadIdx.x >> 5;           // 8 waves
  const int wm = wave & 3, wn = wave >> 2;     // 4x2 wave grid, 32x32 per wave
  const int lo = lane & 15, hi = (lane >> 4) & 1, hi8 = hi * 8;
  const int bm = blockIdx.y * 128;
  const int bn = blockIdx.x * 64;

  int rA[2], cN[2];
#pragma unroll
  for (int sm = 0; sm < 2; ++sm) rA[sm] = bm + wm * 32 + sm * 16 + lo;
#pragma unroll
  for (int sn = 0; sn < 2; ++sn) cN[sn] = bn + wn * 32 + sn * 16 + lo;

  v8f acc[2][2];
#pragma unroll
  for (int sm = 0; sm < 2; ++sm)
#pragma unroll
    for (int sn = 0; sn < 2; ++sn) acc[sm][sn] = zero8();

  for (int kk = 0; kk < K; kk += 32) {
    v16bf afr[2], bfr[2];
#pragma unroll
    for (int sm = 0; sm < 2; ++sm) {
      // A 16x32 bf16 fragment: lane<16 holds K 0-15; lane>=16 holds 16-31
      const bf16* p = A + (size_t)rA[sm] * K + kk + hi * 16;
      afr[sm] = ld_pair(p, p + 8);
    }
#pragma unroll
    for (int sn = 0; sn < 2; ++sn) {
      // B 32x16 fragment = column n of W.T = row n of W: 16 contiguous K per lane
      const bf16* p = W + (size_t)cN[sn] * K + kk + hi * 16;
      bfr[sn] = ld_pair(p, p + 8);
    }
    if (kk + 32 < K) {   // gfx1250 global_prefetch_b8 for next K step
      __builtin_prefetch(A + (size_t)rA[0] * K + kk + 32, 0, 1);
      __builtin_prefetch(W + (size_t)cN[0] * K + kk + 32, 0, 1);
    }
#pragma unroll
    for (int sm = 0; sm < 2; ++sm)
#pragma unroll
      for (int sn = 0; sn < 2; ++sn)
        acc[sm][sn] = wmma_bf16(afr[sm], bfr[sn], acc[sm][sn]);
  }

#pragma unroll
  for (int sm = 0; sm < 2; ++sm) {
#pragma unroll
    for (int sn = 0; sn < 2; ++sn) {
      const int n = cN[sn];
      const float bsn = bias[n];
#pragma unroll
      for (int j = 0; j < 8; ++j) {
        const int row = bm + wm * 32 + sm * 16 + j + hi8;
        float val = (acc[sm][sn][j] + bsn) * alpha;
        if constexpr (MODE == 0) {
          Cout[(size_t)row * N + n] = (OutT)val;
        } else {
          const int bb = row / Stok;
          const int t  = row - bb * Stok;
          ((bf16*)Cout)[((size_t)bb * N + n) * Stok + t] = (bf16)val;
        }
      }
    }
  }
}

// ---------------- fused attention ----------------
// grid: B * S/16 blocks, 512 threads (16 waves; wave == head)
// Q,K: bf16 [B,S,D] (Q pre-scaled by 1/8); Vt: bf16 [B,D,S] channel-major
__global__ __launch_bounds__(512) void attn_fused_kernel(
    const bf16* __restrict__ Qb, const bf16* __restrict__ Kb,
    const bf16* __restrict__ Vt, float* __restrict__ osum,
    float* __restrict__ ocnt, bf16* __restrict__ ctx) {
  __shared__ float s_sum[16 * CHUNK];
  __shared__ float s_cnt[16 * CHUNK];
  __shared__ __align__(16) bf16 s_p[NB_H][16 * 32];

  const int tid  = threadIdx.x;
  const int wave = tid >> 5, lane = tid & 31;
  const int lo = lane & 15, hi = (lane >> 4) & 1, hi8 = hi * 8;
  const int b  = blockIdx.x / (NB_S / 16);
  const int qt = blockIdx.x % (NB_S / 16);
  const int h  = wave;

  const bf16* Qh = Qb + (size_t)b * NB_S * NB_D + (size_t)h * NB_DK;
  const bf16* Kh = Kb + (size_t)b * NB_S * NB_D + (size_t)h * NB_DK;

  // Q fragment (16 rows x 64 d) held for the whole kernel: two 16x32 A-frags
  v16bf aq[2];
#pragma unroll
  for (int d2 = 0; d2 < 2; ++d2) {
    const bf16* qp = Qh + (size_t)(qt * 16 + lo) * NB_D + d2 * 32 + hi * 16;
    aq[d2] = ld_pair(qp, qp + 8);
  }

  // ---- pass 1: per-lane online (max, expsum) over this lane's columns; no shuffles
  float rm[8], rs[8];
#pragma unroll
  for (int j = 0; j < 8; ++j) { rm[j] = -3.0e38f; rs[j] = 0.0f; }

  for (int kt = 0; kt < NB_S / 16; ++kt) {
    v8f s = zero8();
#pragma unroll
    for (int d2 = 0; d2 < 2; ++d2) {
      const bf16* kp = Kh + (size_t)(kt * 16 + lo) * NB_D + d2 * 32 + hi * 16;
      v16bf bk = ld_pair(kp, kp + 8);
      s = wmma_bf16(aq[d2], bk, s);
    }
#pragma unroll
    for (int j = 0; j < 8; ++j) {
      const float nm = fmaxf(rm[j], s[j]);
      rs[j] = rs[j] * __expf(rm[j] - nm) + __expf(s[j] - nm);
      rm[j] = nm;
    }
  }
  // single cross-lane merge of (max, sum) pairs over the 16 columns-lanes
#pragma unroll
  for (int j = 0; j < 8; ++j) {
#pragma unroll
    for (int off = 1; off <= 8; off <<= 1) {
      const float om = __shfl_xor(rm[j], off, 32);
      const float os = __shfl_xor(rs[j], off, 32);
      const float nm = fmaxf(rm[j], om);
      rs[j] = rs[j] * __expf(rm[j] - nm) + os * __expf(om - nm);
      rm[j] = nm;
    }
    rs[j] = 1.0f / rs[j];  // now 1/sum, broadcast in all lanes
  }

  // ---- pass 2: probs, threshold, head-reduced sums in LDS, ctx via WMMA ----
  v8f ctxacc[4];
#pragma unroll
  for (int jd = 0; jd < 4; ++jd) ctxacc[jd] = zero8();

  for (int c = 0; c < NB_S / CHUNK; ++c) {
    for (int i = tid; i < 16 * CHUNK; i += 512) { s_sum[i] = 0.0f; s_cnt[i] = 0.0f; }
    __syncthreads();

    for (int kb2 = 0; kb2 < CHUNK; kb2 += 32) {
      const int kbase = c * CHUNK + kb2;
#pragma unroll
      for (int t2 = 0; t2 < 2; ++t2) {
        const int kt16 = kbase + t2 * 16;
        v8f s = zero8();
#pragma unroll
        for (int d2 = 0; d2 < 2; ++d2) {
          const bf16* kp = Kh + (size_t)(kt16 + lo) * NB_D + d2 * 32 + hi * 16;
          v16bf bk = ld_pair(kp, kp + 8);
          s = wmma_bf16(aq[d2], bk, s);
        }
#pragma unroll
        for (int j = 0; j < 8; ++j) {
          const float p    = __expf(s[j] - rm[j]) * rs[j];
          const float kept = (p >= 0.1f) ? 1.0f : 0.0f;
          const int row = j + hi8;
          const int col = kb2 + t2 * 16 + lo;
          atomicAdd(&s_sum[row * CHUNK + col], p);       // sum over heads (ds_add_f32)
          atomicAdd(&s_cnt[row * CHUNK + col], kept);    // mask count over heads
          s_p[wave][row * 32 + t2 * 16 + lo] = (bf16)(p * kept);
        }
      }
      // masked probs (16x32) -> A-fragment, ctx += P @ V
      const bf16* pp = &s_p[wave][lo * 32 + hi * 16];
      v16bf pa = ld_pair(pp, pp + 8);
#pragma unroll
      for (int jd = 0; jd < 4; ++jd) {
        // V channel-major: column d holds S contiguous keys
        const bf16* vcol =
            Vt + ((size_t)b * NB_D + h * NB_DK + jd * 16 + lo) * NB_S + kbase + hi * 16;
        v16bf bv = ld_pair(vcol, vcol + 8);
        ctxacc[jd] = wmma_bf16(pa, bv, ctxacc[jd]);
      }
    }
    __syncthreads();

    // cooperative write of head-reduced sum/mask chunk
    const size_t base = ((size_t)b * NB_S + qt * 16) * NB_S + (size_t)c * CHUNK;
    for (int i = tid; i < 16 * CHUNK; i += 512) {
      const int r = i / CHUNK, cc = i - r * CHUNK;
      osum[base + (size_t)r * NB_S + cc] = s_sum[i];
      ocnt[base + (size_t)r * NB_S + cc] = s_cnt[i];
    }
    __syncthreads();
  }

  // ---- write context (merged [B,S,D] layout, bf16) ----
#pragma unroll
  for (int jd = 0; jd < 4; ++jd) {
#pragma unroll
    for (int j = 0; j < 8; ++j) {
      const int m = j + hi8;
      const int d = jd * 16 + lo;
      ctx[((size_t)b * NB_S + qt * 16 + m) * NB_D + h * NB_DK + d] = (bf16)ctxacc[jd][j];
    }
  }
}

// ---------------- host launcher ----------------
extern "C" void kernel_launch(void* const* d_in, const int* in_sizes, int n_in,
                              void* d_out, int out_size, void* d_ws, size_t ws_size,
                              hipStream_t stream) {
  const float* x  = (const float*)d_in[0];
  const float* Wq = (const float*)d_in[1];
  const float* bq = (const float*)d_in[2];
  const float* Wk = (const float*)d_in[3];
  const float* bk = (const float*)d_in[4];
  const float* Wv = (const float*)d_in[5];
  const float* bv = (const float*)d_in[6];
  const float* Wo = (const float*)d_in[7];
  const float* bo = (const float*)d_in[8];

  const int M = NB_B * NB_S;      // 4096 tokens
  const int XE = M * NB_D;        // 4,194,304
  const int WE = NB_D * NB_D;     // 1,048,576

  char* w = (char*)d_ws;
  auto carve = [&](size_t bytes) {
    void* r = (void*)w;
    w += (bytes + 255) & ~(size_t)255;
    return r;
  };
  bf16* xb   = (bf16*)carve((size_t)XE * 2);
  bf16* wqb  = (bf16*)carve((size_t)WE * 2);
  bf16* wkb  = (bf16*)carve((size_t)WE * 2);
  bf16* wvb  = (bf16*)carve((size_t)WE * 2);
  bf16* wob  = (bf16*)carve((size_t)WE * 2);
  bf16* qbuf = (bf16*)carve((size_t)XE * 2);
  bf16* kbuf = (bf16*)carve((size_t)XE * 2);
  bf16* vtb  = (bf16*)carve((size_t)XE * 2);  // [B, D, S] channel-major
  bf16* ctxb = (bf16*)carve((size_t)XE * 2);

  float* out_proj = (float*)d_out;                       // [B,S,D]
  float* out_sum  = out_proj + (size_t)XE;               // [B,S,S]
  float* out_cnt  = out_sum + (size_t)NB_B * NB_S * NB_S;

  // 1) convert inputs to bf16
  cvt_f32_to_bf16<<<XE / 1024, 256, 0, stream>>>(x, xb, XE);
  cvt_f32_to_bf16<<<WE / 1024, 256, 0, stream>>>(Wq, wqb, WE);
  cvt_f32_to_bf16<<<WE / 1024, 256, 0, stream>>>(Wk, wkb, WE);
  cvt_f32_to_bf16<<<WE / 1024, 256, 0, stream>>>(Wv, wvb, WE);
  cvt_f32_to_bf16<<<WE / 1024, 256, 0, stream>>>(Wo, wob, WE);

  // 2) projections (WMMA bf16). Q pre-scaled by 1/sqrt(d_k)=0.125; V stored transposed.
  dim3 ggrid(NB_D / 64, M / 128);
  gemm_wmma_kernel<0, bf16><<<ggrid, 256, 0, stream>>>(xb, wqb, bq, qbuf, M, NB_D, NB_D, NB_S, 0.125f);
  gemm_wmma_kernel<0, bf16><<<ggrid, 256, 0, stream>>>(xb, wkb, bk, kbuf, M, NB_D, NB_D, NB_S, 1.0f);
  gemm_wmma_kernel<1, bf16><<<ggrid, 256, 0, stream>>>(xb, wvb, bv, vtb,  M, NB_D, NB_D, NB_S, 1.0f);

  // 3) fused attention: scores->softmax->threshold->head-reduced sums + ctx
  attn_fused_kernel<<<NB_B * (NB_S / 16), 512, 0, stream>>>(qbuf, kbuf, vtb, out_sum,
                                                            out_cnt, ctxb);

  // 4) output projection, fp32 result
  gemm_wmma_kernel<0, float><<<ggrid, 256, 0, stream>>>(ctxb, wob, bo, out_proj, M, NB_D,
                                                        NB_D, NB_S, 1.0f);
}
